// SelfAttention_58875411694140
// MI455X (gfx1250) — matrix operand order repaired and checked by reading, not compiled, not run on hardware
//
#include <hip/hip_runtime.h>
#include <hip/hip_bf16.h>
#include <stdint.h>

// ---------------------------------------------------------------------------
// Shapes (fixed by the reference): B=16, T=1024, C=768, H=12, D=64
// ---------------------------------------------------------------------------
constexpr int Bsz = 16;
constexpr int Tsz = 1024;
constexpr int Csz = 768;
constexpr int Hsz = 12;
constexpr int Dsz = 64;
constexpr int Msz = Bsz * Tsz;          // 16384 rows
constexpr int QKVC = 3 * Csz;           // 2304

typedef __attribute__((ext_vector_type(16))) __bf16 v16bf;
typedef __attribute__((ext_vector_type(8)))  float  v8f;

union FragAB {
    uint4 q[2];   // 32 bytes = 16 bf16 elems = 8 VGPRs
    v16bf v;
};

__device__ __forceinline__ unsigned short f2bf(float f) {
    unsigned u = __float_as_uint(f);
    u += 0x7FFFu + ((u >> 16) & 1u);          // round-nearest-even
    return (unsigned short)(u >> 16);
}
__device__ __forceinline__ unsigned pack2bf(float a, float b) {
    return (unsigned)f2bf(a) | ((unsigned)f2bf(b) << 16);
}

// gfx1250 async global->LDS dword copy (ASYNCcnt-tracked), GVS addressing:
// mem = SADDR(64b sgpr) + VADDR(32b unsigned byte offset); VDST holds the
// wave-relative LDS byte address.
__device__ __forceinline__ void async_g2l_b32(unsigned lds_addr, unsigned voff,
                                              unsigned long long saddr) {
    asm volatile("global_load_async_to_lds_b32 %0, %1, %2"
                 :: "v"(lds_addr), "v"(voff), "s"(saddr)
                 : "memory");
}
__device__ __forceinline__ void wait_asynccnt0() {
    asm volatile("s_wait_asynccnt 0x0" ::: "memory");
}

// ---------------------------------------------------------------------------
// GEMM: out[M,N] = A[M,K] (f32) @ W[K,N] (f32) + bias[N]
// 128x128 tile / workgroup, BK=32, 8 waves each owning a 32x64 region.
// A staged in LDS as bf16 [M][K] (row-major), W staged as bf16 [N][K]
// (transposed) so both fragment types are two ds_load_b128's.
// ---------------------------------------------------------------------------
template <bool OUT_BF16>
__global__ __launch_bounds__(256) void gemm_bias(
    const float* __restrict__ A, const float* __restrict__ W,
    const float* __restrict__ bias, void* __restrict__ out,
    int M, int N, int K)
{
    __shared__ unsigned short Alds[128 * 40];   // stride 40 elems = 80B (16B aligned)
    __shared__ unsigned short Wlds[128 * 40];   // [n][k] transposed

    const int tid  = threadIdx.x;
    const int lane = tid & 31;
    const int wv   = tid >> 5;
    const int half = lane >> 4;
    const int ln   = lane & 15;

    const int tileM = blockIdx.y * 128;
    const int tileN = blockIdx.x * 128;
    const int waveM = (wv & 3) * 32;    // 4 waves along M
    const int waveN = (wv >> 2) * 64;   // 2 waves along N

    v8f acc[2][4];
#pragma unroll
    for (int i = 0; i < 2; ++i)
#pragma unroll
        for (int j = 0; j < 4; ++j)
#pragma unroll
            for (int e = 0; e < 8; ++e) acc[i][j][e] = 0.0f;

    unsigned* Ald = (unsigned*)Alds;
    unsigned* Wld = (unsigned*)Wlds;

    for (int kk = 0; kk < K; kk += 32) {
        // prefetch next K-slice while this one is being consumed
        if (kk + 32 < K) {
            __builtin_prefetch(A + (size_t)(tileM + (tid >> 3)) * K + kk + 32 + (tid & 7) * 4, 0, 3);
            __builtin_prefetch(W + (size_t)(kk + 32 + (tid >> 7)) * N + tileN + (tid & 127), 0, 3);
        }
        __syncthreads();
        // ---- stage A tile 128x32 f32 -> bf16 LDS (float4 coalesced) ----
        {
            const int mrow = tid >> 3;          // 0..31
            const int c4   = (tid & 7) * 4;     // 0..28
#pragma unroll
            for (int p = 0; p < 4; ++p) {
                const int m = mrow + p * 32;
                const float4 f = *(const float4*)(A + (size_t)(tileM + m) * K + kk + c4);
                Ald[m * 20 + (c4 >> 1)]     = pack2bf(f.x, f.y);
                Ald[m * 20 + (c4 >> 1) + 1] = pack2bf(f.z, f.w);
            }
        }
        // ---- stage W tile 32x128 f32 -> bf16 LDS transposed [n][k] ----
        {
            const int n  = tid & 127;
            const int j0 = tid >> 7;            // 0..1
#pragma unroll
            for (int p = 0; p < 8; ++p) {
                const int j = j0 + p * 2;       // k-pair index 0..15
                const float w0 = W[(size_t)(kk + 2 * j)     * N + tileN + n];
                const float w1 = W[(size_t)(kk + 2 * j + 1) * N + tileN + n];
                Wld[n * 20 + j] = pack2bf(w0, w1);
            }
        }
        __syncthreads();

        // ---- fragments + 8 WMMAs per wave ----
        FragAB af[2];
#pragma unroll
        for (int i = 0; i < 2; ++i) {
            const int m = waveM + i * 16 + ln;
            const char* base = (const char*)Alds + m * 80 + half * 16;
            af[i].q[0] = *(const uint4*)(base);
            af[i].q[1] = *(const uint4*)(base + 32);
        }
#pragma unroll
        for (int j = 0; j < 4; ++j) {
            const int n = waveN + j * 16 + ln;
            FragAB bf;
            const char* base = (const char*)Wlds + n * 80 + half * 16;
            bf.q[0] = *(const uint4*)(base);
            bf.q[1] = *(const uint4*)(base + 32);
#pragma unroll
            for (int i = 0; i < 2; ++i)
                acc[i][j] = __builtin_amdgcn_wmma_f32_16x16x32_bf16(
                    false, af[i].v, false, bf.v, (short)0, acc[i][j], false, false);
        }
    }

    // ---- epilogue: bias + store ----
    float bc[4];
#pragma unroll
    for (int j = 0; j < 4; ++j) bc[j] = bias[tileN + waveN + j * 16 + ln];

#pragma unroll
    for (int i = 0; i < 2; ++i)
#pragma unroll
        for (int r = 0; r < 8; ++r) {
            const int row = tileM + waveM + i * 16 + r + (half ? 8 : 0);
#pragma unroll
            for (int j = 0; j < 4; ++j) {
                const int col = tileN + waveN + j * 16 + ln;
                const float val = acc[i][j][r] + bc[j];
                if (OUT_BF16)
                    ((unsigned short*)out)[(size_t)row * N + col] = f2bf(val);
                else
                    ((float*)out)[(size_t)row * N + col] = val;
            }
        }
}

// ---------------------------------------------------------------------------
// Flash attention (causal). qkv is bf16 [B*T][3C]; per WG: one (b,h) and one
// 64-row query block; 4 waves, 16 query rows each. K tile staged [key][d]
// via async global->LDS DMA; V staged transposed [d][key] with packed dword
// stores; Q fragments loaded straight from global (the A-fragment pattern is
// two aligned b128 loads per k-chunk).
// ---------------------------------------------------------------------------
__global__ __launch_bounds__(128) void attn_fwd(
    const unsigned short* __restrict__ qkv, float* __restrict__ y)
{
    __shared__ unsigned short Klds[64 * 72];     // [key][d], stride 72 (144B)
    __shared__ unsigned short Vt[64 * 72];       // [d][key]
    __shared__ unsigned short Pb[4 * 16 * 72];   // per-wave P bounce buffer

    const int tid  = threadIdx.x;
    const int lane = tid & 31;
    const int wv   = tid >> 5;
    const int half = lane >> 4;
    const int ln   = lane & 15;

    const int b  = blockIdx.y / Hsz;
    const int h  = blockIdx.y % Hsz;
    const int qb = blockIdx.x;
    const int q0 = qb * 64;
    const int qrow0 = q0 + wv * 16;

    const unsigned short* qbase = qkv + ((size_t)b * Tsz) * QKVC + h * Dsz;
    const unsigned short* kbase = qbase + Csz;
    const unsigned short* vbase = qbase + 2 * Csz;

    // Q fragments (resident for whole kernel)
    FragAB qf[2];
    {
        const unsigned short* rowp = qbase + (size_t)(qrow0 + ln) * QKVC;
#pragma unroll
        for (int kc = 0; kc < 2; ++kc) {
            const char* base = (const char*)rowp + kc * 64 + half * 16;
            qf[kc].q[0] = *(const uint4*)(base);
            qf[kc].q[1] = *(const uint4*)(base + 32);
        }
    }

    float mrow[8], lrow[8];
    v8f oacc[4];
#pragma unroll
    for (int r = 0; r < 8; ++r) { mrow[r] = -3.0e38f; lrow[r] = 0.0f; }
#pragma unroll
    for (int dn = 0; dn < 4; ++dn)
#pragma unroll
        for (int e = 0; e < 8; ++e) oacc[dn][e] = 0.0f;

    const unsigned kldsbase = (unsigned)(size_t)(void*)Klds;   // LDS byte offset
    unsigned* vt32 = (unsigned*)Vt;
    const int pbase = wv * 16 * 72;

    for (int kb = 0; kb <= qb; ++kb) {
        const int key0 = kb * 64;
        __syncthreads();

        // ---- stage K [key][d] via async DMA: 64 rows x 32 dwords ----
        {
            const unsigned long long ksaddr =
                (unsigned long long)(uintptr_t)(kbase + (size_t)key0 * QKVC);
#pragma unroll
            for (int p = 0; p < 16; ++p) {
                const int idx = tid + p * 128;
                const int key = idx >> 5;       // 0..63
                const int c2  = idx & 31;       // dword column (2 bf16)
                async_g2l_b32(kldsbase + key * 144 + c2 * 4,
                              (unsigned)(key * (QKVC * 2) + c2 * 4), ksaddr);
            }
        }
        // ---- stage V transposed [d][key] with packed dword stores ----
#pragma unroll
        for (int p = 0; p < 8; ++p) {
            const int idx = tid + p * 128;
            const int key = (idx >> 5) * 2;     // even key of a pair
            const int c2  = idx & 31;           // dword column (d = 2*c2, 2*c2+1)
            const unsigned v0 = *(const unsigned*)(vbase + (size_t)(key0 + key)     * QKVC + c2 * 2);
            const unsigned v1 = *(const unsigned*)(vbase + (size_t)(key0 + key + 1) * QKVC + c2 * 2);
            vt32[((2 * c2)     * 72 + key) >> 1] = (v0 & 0xffffu) | (v1 << 16);
            vt32[((2 * c2 + 1) * 72 + key) >> 1] = (v0 >> 16) | (v1 & 0xffff0000u);
        }
        wait_asynccnt0();
        __syncthreads();

        // ---- S = (Q K^T) * scale, causal mask ----
        v8f sacc[4];
#pragma unroll
        for (int n = 0; n < 4; ++n)
#pragma unroll
            for (int e = 0; e < 8; ++e) sacc[n][e] = 0.0f;

#pragma unroll
        for (int n = 0; n < 4; ++n)
#pragma unroll
            for (int kc = 0; kc < 2; ++kc) {
                FragAB bfK;
                const char* base = (const char*)Klds + (n * 16 + ln) * 144 + kc * 64 + half * 16;
                bfK.q[0] = *(const uint4*)(base);
                bfK.q[1] = *(const uint4*)(base + 32);
                sacc[n] = __builtin_amdgcn_wmma_f32_16x16x32_bf16(
                    false, qf[kc].v, false, bfK.v, (short)0, sacc[n], false, false);
            }

        // ---- online softmax (row M=r in lanes 0-15, M=r+8 in lanes 16-31) ----
#pragma unroll
        for (int r = 0; r < 8; ++r) {
            const int qi = qrow0 + r + (half ? 8 : 0);
            float rmax = -3.0e38f;
#pragma unroll
            for (int n = 0; n < 4; ++n) {
                float s = sacc[n][r] * 0.125f;            // 1/sqrt(64)
                const int key = key0 + n * 16 + ln;
                s = (key > qi) ? -3.0e38f : s;
                sacc[n][r] = s;
                rmax = fmaxf(rmax, s);
            }
#pragma unroll
            for (int d = 1; d < 16; d <<= 1)
                rmax = fmaxf(rmax, __shfl_xor(rmax, d, 16));
            const float newm  = fmaxf(mrow[r], rmax);
            const float alpha = __expf(mrow[r] - newm);
            mrow[r] = newm;

            float rsum = 0.0f;
            const int m = r + (half ? 8 : 0);
#pragma unroll
            for (int n = 0; n < 4; ++n) {
                const float pe = __expf(sacc[n][r] - newm);
                rsum += pe;
                Pb[pbase + m * 72 + n * 16 + ln] = f2bf(pe);   // C-layout -> A-layout bounce
            }
#pragma unroll
            for (int d = 1; d < 16; d <<= 1)
                rsum += __shfl_xor(rsum, d, 16);
            lrow[r] = lrow[r] * alpha + rsum;
#pragma unroll
            for (int dn = 0; dn < 4; ++dn) oacc[dn][r] *= alpha;
        }

        // ---- O += P @ V (DS ops per-wave are in order: bounce is safe) ----
        FragAB paf[2];
#pragma unroll
        for (int kc = 0; kc < 2; ++kc) {
            const char* base = (const char*)Pb + (pbase + ln * 72) * 2 + kc * 64 + half * 16;
            paf[kc].q[0] = *(const uint4*)(base);
            paf[kc].q[1] = *(const uint4*)(base + 32);
        }
#pragma unroll
        for (int dn = 0; dn < 4; ++dn)
#pragma unroll
            for (int kc = 0; kc < 2; ++kc) {
                FragAB vf;
                const char* base = (const char*)Vt + (dn * 16 + ln) * 144 + kc * 64 + half * 16;
                vf.q[0] = *(const uint4*)(base);
                vf.q[1] = *(const uint4*)(base + 32);
                oacc[dn] = __builtin_amdgcn_wmma_f32_16x16x32_bf16(
                    false, paf[kc].v, false, vf.v, (short)0, oacc[dn], false, false);
            }
    }

    // ---- normalize and write y[b][t][h*64+d] (f32) ----
#pragma unroll
    for (int r = 0; r < 8; ++r) {
        const float inv = 1.0f / lrow[r];
        const int t = qrow0 + r + (half ? 8 : 0);
        float* yrow = y + ((size_t)b * Tsz + t) * Csz + h * Dsz;
#pragma unroll
        for (int dn = 0; dn < 4; ++dn)
            yrow[dn * 16 + ln] = oacc[dn][r] * inv;
    }
}

// ---------------------------------------------------------------------------
// Launcher: x -> (GEMM, bf16 qkv in ws) -> flash attention (f32 y in ws)
//           -> (GEMM, f32 out). Workspace use: 75.5MB + 48MB.
// ---------------------------------------------------------------------------
extern "C" void kernel_launch(void* const* d_in, const int* in_sizes, int n_in,
                              void* d_out, int out_size, void* d_ws, size_t ws_size,
                              hipStream_t stream) {
    (void)in_sizes; (void)n_in; (void)out_size; (void)ws_size;
    const float* x      = (const float*)d_in[0];
    const float* w_attn = (const float*)d_in[1];
    const float* b_attn = (const float*)d_in[2];
    const float* w_proj = (const float*)d_in[3];
    const float* b_proj = (const float*)d_in[4];
    float* out = (float*)d_out;

    unsigned short* qkv = (unsigned short*)d_ws;                 // bf16 [M][2304]
    const size_t qkv_bytes = (size_t)Msz * QKVC * sizeof(unsigned short);
    float* yattn = (float*)((char*)d_ws + ((qkv_bytes + 255) & ~(size_t)255));

    // 1) qkv = x @ w_attn + b_attn  (store bf16)
    gemm_bias<true><<<dim3(QKVC / 128, Msz / 128), dim3(256), 0, stream>>>(
        x, w_attn, b_attn, qkv, Msz, QKVC, Csz);

    // 2) flash attention
    attn_fwd<<<dim3(Tsz / 64, Bsz * Hsz), dim3(128), 0, stream>>>(qkv, yattn);

    // 3) out = y @ w_proj + b_proj  (store f32)
    gemm_bias<false><<<dim3(Csz / 128, Msz / 128), dim3(256), 0, stream>>>(
        yattn, w_proj, b_proj, out, Msz, Csz, Csz);
}
